// PredictionNetwork_27479200760525
// MI455X (gfx1250) — compile-verified, matrix operand order checked by reading, not run on hardware
//
#include <hip/hip_runtime.h>

typedef __attribute__((ext_vector_type(16))) _Float16 v16h;
typedef __attribute__((ext_vector_type(8)))  float    v8f;

#define NB     256      // batch
#define NI     10240    // seq*F inputs to cell A
#define NU     32       // cell A units
#define BT     16       // batch rows per tile (WMMA M)
#define NBT    (NB/BT)  // 16 b-tiles
#define IPB    256      // i per block window
#define NWGI   (NI/IPB) // 40 i-windows
#define IPW    64       // i per wave (2 chunks of 32)
#define NSL    NWGI     // partial slices per (u,b)
#define UNF    6

__device__ __forceinline__ float fast_sigmoid(float t) {
    // 1/(1+exp(-t));  v_exp_f32 + v_rcp_f32 (trans pipe)
    float e = __expf(-t);
    return __builtin_amdgcn_rcpf(1.0f + e);
}

// ---------------------------------------------------------------------------
// Kernel 1: sensory synapse stage of cell A.
// A = (sact*serev) tile (f16, 16x32), B = ones in column 0 only (constant!).
// D += A*B via v_wmma_f32_16x16x32_f16 -> column 0 = sum_i sact*serev.
// den = sum_i sact accumulated on the VALU in f32 (1 shfl_xor per u).
// ---------------------------------------------------------------------------
__global__ __launch_bounds__(128)
void ltc_sensory_wmma(const float* __restrict__ x,
                      const float* __restrict__ a_iw,
                      const float* __restrict__ a_ib,
                      const float* __restrict__ a_smu,
                      const float* __restrict__ a_ssig,
                      const float* __restrict__ a_sW,
                      const float* __restrict__ a_serev,
                      float* __restrict__ P_num,
                      float* __restrict__ P_den) {
    __shared__ float xs[IPB][20];       // [i_local][b_local], pad 20: conflict-free A reads
    __shared__ float acc[4 * NU * BT * 2];

    const int tid   = threadIdx.x;
    const int bid   = blockIdx.x;
    const int btile = bid & (NBT - 1);  // 0..15
    const int wgi   = bid >> 4;         // 0..39
    const int ibase = wgi * IPB;
    const int b0    = btile * BT;

    // Stage x tile (affine input mapping applied), transposed for lane access.
    for (int bl = 0; bl < BT; ++bl) {
        const float* xrow = x + (size_t)(b0 + bl) * NI + ibase;
        #pragma unroll
        for (int rep = 0; rep < 2; ++rep) {
            int il = rep * 128 + tid;
            int ig = ibase + il;
            xs[il][bl] = xrow[il] * a_iw[ig] + a_ib[ig];
        }
    }
    // Pull the parameter stream for this i-window toward the caches (gfx1250
    // global_prefetch_b8); one line per even i-row (32 floats = 128B).
    {
        int ig = ibase + tid * 2;
        __builtin_prefetch(a_smu   + (size_t)ig * NU, 0, 1);
        __builtin_prefetch(a_ssig  + (size_t)ig * NU, 0, 1);
        __builtin_prefetch(a_sW    + (size_t)ig * NU, 0, 1);
        __builtin_prefetch(a_serev + (size_t)ig * NU, 0, 1);
    }
    __syncthreads();

    const int w    = tid >> 5;          // wave in block: i-slice owner
    const int lane = tid & 31;
    const int m    = lane & 15;         // A row (batch) / B,D column
    const int half = lane >> 4;
    const int iw0  = w * IPW;           // wave's i-slice start (local)

    // Constant B: lanes with N==0 (lanes 0 and 16) hold 1.0 in all K slots.
    // Any K permutation of a constant column is identical -> layout-proof.
    v16h Bm;
    {
        const _Float16 bv = (m == 0) ? (_Float16)1.0f : (_Float16)0.0f;
        #pragma unroll
        for (int s = 0; s < 16; ++s) Bm[s] = bv;
    }

    for (int u = 0; u < NU; ++u) {
        v8f c = {0.f, 0.f, 0.f, 0.f, 0.f, 0.f, 0.f, 0.f};
        float dsum = 0.f;               // f32 den partial (this lane's 16 K slots)
        #pragma unroll
        for (int ch = 0; ch < 2; ++ch) {
            const int lc = iw0 + ch * 32;   // chunk start, local i
            const int gc = ibase + lc;      // chunk start, global i
            v16h A;
            #pragma unroll
            for (int s = 0; s < 16; ++s) {
                // A (16x32 f16): lane L row M=L%16; K = s + 8*half + 8*(s>=8)
                const int il = s + half * 8 + ((s >= 8) ? 8 : 0);
                const int gi = gc + il;
                const size_t p = (size_t)gi * NU + u;   // shared index, 4 streams
                const float xv  = xs[lc + il][m];
                const float smu = a_smu [p];
                const float ssg = a_ssig[p];
                const float sw  = a_sW  [p];
                const float se  = a_serev[p];           // +-1
                const float sv  = sw * fast_sigmoid((xv - smu) * ssg);
                dsum += sv;
                A[s] = (_Float16)(sv * se);
            }
            c = __builtin_amdgcn_wmma_f32_16x16x32_f16(
                    false, A, false, Bm, (short)0, c, false, false);
        }
        // num: D[row][0]; lane L holds col N=L%16, rows half*8..half*8+7.
        if (m == 0) {
            const int rbase = half * 8;
            #pragma unroll
            for (int r = 0; r < 8; ++r)
                acc[w * 1024 + u * 32 + (rbase + r) * 2 + 0] = c[r];
        }
        // den: merge the two lane-halves (K 0..7,16..23 + K 8..15,24..31 for batch row m).
        const float dtot = dsum + __shfl_xor(dsum, 16, 32);
        if (half == 0)
            acc[w * 1024 + u * 32 + m * 2 + 1] = dtot;
    }
    __syncthreads();

    // Combine the 4 waves' partials, emit per-window slice (deterministic).
    for (int e = tid; e < 1024; e += 128) {
        float vsum = acc[e] + acc[e + 1024] + acc[e + 2048] + acc[e + 3072];
        const int u  = e >> 5;
        const int bl = (e & 31) >> 1;
        const int nd = e & 1;
        float* dst = nd ? P_den : P_num;
        dst[((size_t)u * NB + (b0 + bl)) * NSL + wgi] = vsum;
    }
}

// ---------------------------------------------------------------------------
// Kernel 2: slice reduction + cell A unfolds + cell B + final sigmoid.
// One wave (32 lanes = 32 units) per batch element.
// ---------------------------------------------------------------------------
__global__ __launch_bounds__(32)
void ltc_recurrent(const float* __restrict__ P_num,
                   const float* __restrict__ P_den,
                   const float* __restrict__ a_mu,
                   const float* __restrict__ a_sig,
                   const float* __restrict__ a_W,
                   const float* __restrict__ a_erev,
                   const float* __restrict__ a_gleak,
                   const float* __restrict__ a_vleak,
                   const float* __restrict__ a_cm,
                   const float* __restrict__ b_iw,
                   const float* __restrict__ b_ib,
                   const float* __restrict__ b_smu,
                   const float* __restrict__ b_ssig,
                   const float* __restrict__ b_sW,
                   const float* __restrict__ b_serev,
                   const float* __restrict__ b_mu,
                   const float* __restrict__ b_sig,
                   const float* __restrict__ b_W,
                   const float* __restrict__ b_erev,
                   const float* __restrict__ b_gleak,
                   const float* __restrict__ b_vleak,
                   const float* __restrict__ b_cm,
                   float* __restrict__ out) {
    const int b = blockIdx.x;
    const int j = threadIdx.x;      // postsynaptic unit

    // Reduce sensory partials over the 40 i-windows (fixed order).
    float wnum = 0.f, wden = 0.f;
    {
        const float* pn = P_num + ((size_t)j * NB + b) * NSL;
        const float* pd = P_den + ((size_t)j * NB + b) * NSL;
        for (int s = 0; s < NSL; ++s) { wnum += pn[s]; wden += pd[s]; }
    }

    // Cell A: 6 ODE unfolds, 32x32 recurrent synapses.
    const float cmt  = a_cm[j] * (float)UNF;   // cm / (elapsed/unfolds)
    const float gl   = a_gleak[j];
    const float glvl = gl * a_vleak[j];
    float v = 0.f;
    for (int k = 0; k < UNF; ++k) {
        float sn = 0.f, sd = 0.f;
        for (int i = 0; i < NU; ++i) {
            const float vi = __shfl(v, i, 32);
            const float t  = (vi - a_mu[i * NU + j]) * a_sig[i * NU + j];
            const float wv = a_W[i * NU + j] * fast_sigmoid(t);
            sn += wv * a_erev[i * NU + j];
            sd += wv;
        }
        const float num = cmt * v + glvl + sn + wnum;
        const float den = cmt + gl + sd + wden;
        v = num / den;
    }

    // Cell B: 32 -> 1.  Sensory sums via wave reduction.
    const float x2 = v * b_iw[j] + b_ib[j];
    const float s2 = b_sW[j] * fast_sigmoid((x2 - b_smu[j]) * b_ssig[j]);
    float n2 = s2 * b_serev[j];
    float d2 = s2;
    #pragma unroll
    for (int off = 16; off >= 1; off >>= 1) {
        n2 += __shfl_xor(n2, off, 32);
        d2 += __shfl_xor(d2, off, 32);
    }
    const float cmt2  = b_cm[0] * (float)UNF;
    const float gl2   = b_gleak[0];
    const float glvl2 = gl2 * b_vleak[0];
    float v2 = 0.f;
    for (int k = 0; k < UNF; ++k) {
        const float wv  = b_W[0] * fast_sigmoid((v2 - b_mu[0]) * b_sig[0]);
        const float num = cmt2 * v2 + glvl2 + wv * b_erev[0] + n2;
        const float den = cmt2 + gl2 + wv + d2;
        v2 = num / den;
    }
    if (j == 0) out[b] = 1.0f / (1.0f + expf(-v2));
}

extern "C" void kernel_launch(void* const* d_in, const int* in_sizes, int n_in,
                              void* d_out, int out_size, void* d_ws, size_t ws_size,
                              hipStream_t stream) {
    const float* x        = (const float*)d_in[0];
    const float* a_iw     = (const float*)d_in[1];
    const float* a_ib     = (const float*)d_in[2];
    const float* a_smu    = (const float*)d_in[3];
    const float* a_ssig   = (const float*)d_in[4];
    const float* a_sW     = (const float*)d_in[5];
    const float* a_serev  = (const float*)d_in[6];
    const float* a_mu     = (const float*)d_in[7];
    const float* a_sig    = (const float*)d_in[8];
    const float* a_W      = (const float*)d_in[9];
    const float* a_erev   = (const float*)d_in[10];
    const float* a_gleak  = (const float*)d_in[11];
    const float* a_vleak  = (const float*)d_in[12];
    const float* a_cm     = (const float*)d_in[13];
    const float* b_iw     = (const float*)d_in[14];
    const float* b_ib     = (const float*)d_in[15];
    const float* b_smu    = (const float*)d_in[16];
    const float* b_ssig   = (const float*)d_in[17];
    const float* b_sW     = (const float*)d_in[18];
    const float* b_serev  = (const float*)d_in[19];
    const float* b_mu     = (const float*)d_in[20];
    const float* b_sig    = (const float*)d_in[21];
    const float* b_W      = (const float*)d_in[22];
    const float* b_erev   = (const float*)d_in[23];
    const float* b_gleak  = (const float*)d_in[24];
    const float* b_vleak  = (const float*)d_in[25];
    const float* b_cm     = (const float*)d_in[26];
    float* out = (float*)d_out;

    float* P_num = (float*)d_ws;                       // [32][256][40]
    float* P_den = P_num + (size_t)NU * NB * NSL;      // [32][256][40]

    ltc_sensory_wmma<<<dim3(NBT * NWGI), dim3(128), 0, stream>>>(
        x, a_iw, a_ib, a_smu, a_ssig, a_sW, a_serev, P_num, P_den);

    ltc_recurrent<<<dim3(NB), dim3(32), 0, stream>>>(
        P_num, P_den, a_mu, a_sig, a_W, a_erev, a_gleak, a_vleak, a_cm,
        b_iw, b_ib, b_smu, b_ssig, b_sW, b_serev,
        b_mu, b_sig, b_W, b_erev, b_gleak, b_vleak, b_cm, out);
}
